// S5Model_34093450396237
// MI455X (gfx1250) — compile-verified, matrix operand order checked by reading, not run on hardware
//
#include <hip/hip_runtime.h>
#include <math.h>

#define BATCH   8
#define SEQ     2048
#define DIN     64
#define DM      1024
#define DM2     2048
#define NL      4
#define MTOK    (BATCH * SEQ)   // 16384 tokens
#define CH      16              // scan chunks per sequence
#define CLEN    (SEQ / CH)      // 128 steps per chunk

typedef __attribute__((ext_vector_type(16))) _Float16 v16h;
typedef __attribute__((ext_vector_type(8)))  _Float16 v8h;
typedef __attribute__((ext_vector_type(8)))  float    v8f;

union AFrag { v16h v; v8h h[2]; };
union CAcc  { v8f  v; float f[8]; };

__device__ __forceinline__ float gelu_exact(float x) {
    return 0.5f * x * (1.0f + erff(x * 0.7071067811865475f));
}

// ---------------------------------------------------------------------------
// WMMA GEMM:  C[M,N] = A[M,K] * Wt[N,K]^T  (+ bias[N] optional)
// A  : fp16 row-major [M,K]   (activations)
// Wt : fp16 row-major [N,K]   (weight pre-transposed -> same fragment shape)
// C  : fp32 row-major [M,N]
// 256 threads = 8 waves arranged 4(M) x 2(N); each wave owns a 32x64 tile
// (2 A-frags x 4 B-frags -> 8 v_wmma per k-step, 12 b128 loads per k-step).
// Block tile: 128x128.
// ---------------------------------------------------------------------------
__global__ __launch_bounds__(256)
void gemm_f16_wmma(const _Float16* __restrict__ A,
                   const _Float16* __restrict__ Wt,
                   float* __restrict__ C,
                   const float* __restrict__ bias,
                   int M, int N, int K)
{
    const int lane = threadIdx.x & 31;
    const int wave = threadIdx.x >> 5;
    const int m0 = blockIdx.x * 128 + (wave >> 1) * 32;
    const int n0 = blockIdx.y * 128 + (wave & 1) * 64;

    // fragment addressing per CDNA5 16-bit layout:
    // lanes 0-15: row r, K {0..7, 16..23}; lanes 16-31: row r, K {8..15, 24..31}
    const int rr = lane & 15;
    const int kg = (lane >> 4) * 8;

    const _Float16* pa0 = A  + (size_t)(m0 + rr)      * K + kg;
    const _Float16* pa1 = A  + (size_t)(m0 + 16 + rr) * K + kg;
    const _Float16* pb0 = Wt + (size_t)(n0 + rr)      * K + kg;
    const _Float16* pb1 = Wt + (size_t)(n0 + 16 + rr) * K + kg;
    const _Float16* pb2 = Wt + (size_t)(n0 + 32 + rr) * K + kg;
    const _Float16* pb3 = Wt + (size_t)(n0 + 48 + rr) * K + kg;

    CAcc c00 = {}, c01 = {}, c02 = {}, c03 = {};
    CAcc c10 = {}, c11 = {}, c12 = {}, c13 = {};

    for (int k0 = 0; k0 < K; k0 += 32) {
        AFrag a0, a1, b0, b1, b2, b3;
        a0.h[0] = *(const v8h*)(pa0 + k0); a0.h[1] = *(const v8h*)(pa0 + k0 + 16);
        a1.h[0] = *(const v8h*)(pa1 + k0); a1.h[1] = *(const v8h*)(pa1 + k0 + 16);
        b0.h[0] = *(const v8h*)(pb0 + k0); b0.h[1] = *(const v8h*)(pb0 + k0 + 16);
        b1.h[0] = *(const v8h*)(pb1 + k0); b1.h[1] = *(const v8h*)(pb1 + k0 + 16);
        b2.h[0] = *(const v8h*)(pb2 + k0); b2.h[1] = *(const v8h*)(pb2 + k0 + 16);
        b3.h[0] = *(const v8h*)(pb3 + k0); b3.h[1] = *(const v8h*)(pb3 + k0 + 16);
        // group by shared A operand (hardware operand-reuse friendly ordering)
        c00.v = __builtin_amdgcn_wmma_f32_16x16x32_f16(false, a0.v, false, b0.v, (short)0, c00.v, false, false);
        c01.v = __builtin_amdgcn_wmma_f32_16x16x32_f16(false, a0.v, false, b1.v, (short)0, c01.v, false, false);
        c02.v = __builtin_amdgcn_wmma_f32_16x16x32_f16(false, a0.v, false, b2.v, (short)0, c02.v, false, false);
        c03.v = __builtin_amdgcn_wmma_f32_16x16x32_f16(false, a0.v, false, b3.v, (short)0, c03.v, false, false);
        c10.v = __builtin_amdgcn_wmma_f32_16x16x32_f16(false, a1.v, false, b0.v, (short)0, c10.v, false, false);
        c11.v = __builtin_amdgcn_wmma_f32_16x16x32_f16(false, a1.v, false, b1.v, (short)0, c11.v, false, false);
        c12.v = __builtin_amdgcn_wmma_f32_16x16x32_f16(false, a1.v, false, b2.v, (short)0, c12.v, false, false);
        c13.v = __builtin_amdgcn_wmma_f32_16x16x32_f16(false, a1.v, false, b3.v, (short)0, c13.v, false, false);
    }

    // C/D layout: lanes 0-15 -> N=lane, M = r; lanes 16-31 -> N=lane-16, M = 8+r
    const int cn = lane & 15;
    const int cm = (lane >> 4) * 8;
    float bia[4] = {0.0f, 0.0f, 0.0f, 0.0f};
    if (bias) {
        bia[0] = bias[n0 + cn];      bia[1] = bias[n0 + 16 + cn];
        bia[2] = bias[n0 + 32 + cn]; bia[3] = bias[n0 + 48 + cn];
    }
    for (int r = 0; r < 8; ++r) {
        size_t row0 = (size_t)(m0 + cm + r)      * N + n0 + cn;
        size_t row1 = (size_t)(m0 + 16 + cm + r) * N + n0 + cn;
        C[row0]      = c00.f[r] + bia[0];
        C[row0 + 16] = c01.f[r] + bia[1];
        C[row0 + 32] = c02.f[r] + bia[2];
        C[row0 + 48] = c03.f[r] + bia[3];
        C[row1]      = c10.f[r] + bia[0];
        C[row1 + 16] = c11.f[r] + bia[1];
        C[row1 + 32] = c12.f[r] + bia[2];
        C[row1 + 48] = c13.f[r] + bia[3];
    }
}

// ---------------------------------------------------------------------------
// SSM derived parameters: lam_bar = exp(lam*dt);  s = (lam_bar - 1)/lam
// ---------------------------------------------------------------------------
__global__ void ssm_params(const float* __restrict__ lam_re,
                           const float* __restrict__ lam_im,
                           const float* __restrict__ log_step,
                           float* __restrict__ sr, float* __restrict__ si,
                           float* __restrict__ lbr, float* __restrict__ lbi)
{
    int i = blockIdx.x * blockDim.x + threadIdx.x;
    float dt = expf(log_step[i]);
    float lr = lam_re[i], li = lam_im[i];
    float er = expf(lr * dt);
    float br = er * cosf(li * dt);
    float bi = er * sinf(li * dt);
    lbr[i] = br; lbi[i] = bi;
    float nr = br - 1.0f, ni = bi;
    float den = lr * lr + li * li;
    sr[i] = (nr * lr + ni * li) / den;
    si[i] = (ni * lr - nr * li) / den;
}

// B_bar packing: rows [0,DM)=re plane, rows [DM,2DM)=im plane, each [P rows x H cols]
__global__ void pack_bbar(const float* __restrict__ b_re, const float* __restrict__ b_im,
                          const float* __restrict__ sr, const float* __restrict__ si,
                          _Float16* __restrict__ wbu)
{
    int i = blockIdx.x * blockDim.x + threadIdx.x;   // over DM*DM, i = p*DM + h
    int p = i >> 10;
    float re = sr[p] * b_re[i] - si[p] * b_im[i];
    float im = sr[p] * b_im[i] + si[p] * b_re[i];
    wbu[i]                   = (_Float16)re;
    wbu[(size_t)DM * DM + i] = (_Float16)im;
}

// C packing: Wc[h, 0:DM] = c_re[h,:], Wc[h, DM:2DM] = -c_im[h,:]  -> [DM, 2DM]
__global__ void pack_c(const float* __restrict__ c_re, const float* __restrict__ c_im,
                       _Float16* __restrict__ wc)
{
    int i = blockIdx.x * blockDim.x + threadIdx.x;   // i = h*DM + p
    int h = i >> 10, p = i & (DM - 1);
    wc[(size_t)h * DM2 + p]      = (_Float16)c_re[i];
    wc[(size_t)h * DM2 + DM + p] = (_Float16)(-c_im[i]);
}

// generic transpose+convert: w fp32 [K,N] -> wt fp16 [N,K]
__global__ void pack_transpose(const float* __restrict__ w, _Float16* __restrict__ wt,
                               int K, int N)
{
    int i = blockIdx.x * blockDim.x + threadIdx.x;   // over N*K
    int n = i / K, k = i - n * K;
    wt[i] = (_Float16)w[(size_t)k * N + n];
}

__global__ void f32_to_f16(const float* __restrict__ in, _Float16* __restrict__ out, int n)
{
    int i = blockIdx.x * blockDim.x + threadIdx.x;
    if (i < n) out[i] = (_Float16)in[i];
}

// ---------------------------------------------------------------------------
// Chunked complex scan over L, in place on bu [B*L, 2*DM] (col p = re, p+DM = im).
// Pass 1: chunk-local inclusive scans with zero init; emit chunk carries.
// Pass 2: per-chain exclusive prefix over CH chunk carries with W = lam_bar^CLEN.
// Pass 3: x_l += lam_bar^{l+1} * carry_in   for each chunk.
// ---------------------------------------------------------------------------
__global__ void scan_chunk(float* __restrict__ bu,
                           const float* __restrict__ lbr, const float* __restrict__ lbi,
                           float* __restrict__ car_r, float* __restrict__ car_i)
{
    int idx = blockIdx.x * blockDim.x + threadIdx.x;  // over BATCH*CH*DM
    int b = idx >> 14;                 // CH*DM = 16384
    int k = (idx >> 10) & (CH - 1);
    int p = idx & (DM - 1);
    float ar = lbr[p], ai = lbi[p];
    float xr = 0.0f, xi = 0.0f;
    size_t base = ((size_t)b * SEQ + (size_t)k * CLEN) * DM2 + p;
    #pragma unroll 4
    for (int l = 0; l < CLEN; ++l) {
        float ur = bu[base], ui = bu[base + DM];
        float nr = ar * xr - ai * xi + ur;
        float ni = ar * xi + ai * xr + ui;
        xr = nr; xi = ni;
        bu[base] = xr; bu[base + DM] = xi;
        base += DM2;
    }
    car_r[idx] = xr;    // carry layout matches idx = (b*CH + k)*DM + p
    car_i[idx] = xi;
}

__global__ void scan_carry(const float* __restrict__ lbr, const float* __restrict__ lbi,
                           float* __restrict__ car_r, float* __restrict__ car_i)
{
    int idx = blockIdx.x * blockDim.x + threadIdx.x;  // over BATCH*DM
    int b = idx >> 10, p = idx & (DM - 1);
    float ar = lbr[p], ai = lbi[p];
    // W = lam_bar^CLEN via repeated squaring (CLEN = 128 = 2^7)
    float wr = ar, wi = ai;
    #pragma unroll
    for (int t = 0; t < 7; ++t) {
        float nr = wr * wr - wi * wi;
        float ni = 2.0f * wr * wi;
        wr = nr; wi = ni;
    }
    float xr = 0.0f, xi = 0.0f;     // running true end-state X_{k-1}
    for (int k = 0; k < CH; ++k) {
        size_t c = ((size_t)(b * CH + k)) * DM + p;
        float yr = car_r[c], yi = car_i[c];          // chunk-local end state
        car_r[c] = xr; car_i[c] = xi;                // exclusive carry-in for chunk k
        float nr = yr + wr * xr - wi * xi;           // X_k = y_k + W * X_{k-1}
        float ni = yi + wr * xi + wi * xr;
        xr = nr; xi = ni;
    }
}

__global__ void scan_apply(float* __restrict__ bu,
                           const float* __restrict__ lbr, const float* __restrict__ lbi,
                           const float* __restrict__ car_r, const float* __restrict__ car_i)
{
    int idx = blockIdx.x * blockDim.x + threadIdx.x;  // over BATCH*CH*DM
    int b = idx >> 14;
    int k = (idx >> 10) & (CH - 1);
    int p = idx & (DM - 1);
    if (k == 0) return;                                // carry-in is zero
    float cr = car_r[idx], ci = car_i[idx];
    float ar = lbr[p], ai = lbi[p];
    float wr = ar, wi = ai;                            // lam_bar^{l+1}, starts at l=0
    size_t base = ((size_t)b * SEQ + (size_t)k * CLEN) * DM2 + p;
    #pragma unroll 4
    for (int l = 0; l < CLEN; ++l) {
        bu[base]      += wr * cr - wi * ci;
        bu[base + DM] += wr * ci + wi * cr;
        float nr = wr * ar - wi * ai;
        float ni = wr * ai + wi * ar;
        wr = nr; wi = ni;
        base += DM2;
    }
}

// ---------------------------------------------------------------------------
// Row-wise LayerNorm -> fp16 output (GEMM A operand). One block per row.
// ---------------------------------------------------------------------------
__global__ __launch_bounds__(256)
void ln_to_f16(const float* __restrict__ X, const float* __restrict__ g,
               const float* __restrict__ b, _Float16* __restrict__ out)
{
    __shared__ float sred[256];
    const int row = blockIdx.x;
    const float* x = X + (size_t)row * DM;
    float s = 0.0f;
    for (int i = threadIdx.x; i < DM; i += 256) s += x[i];
    sred[threadIdx.x] = s; __syncthreads();
    for (int w = 128; w > 0; w >>= 1) { if (threadIdx.x < w) sred[threadIdx.x] += sred[threadIdx.x + w]; __syncthreads(); }
    float mean = sred[0] * (1.0f / DM);
    __syncthreads();
    float v = 0.0f;
    for (int i = threadIdx.x; i < DM; i += 256) { float d = x[i] - mean; v += d * d; }
    sred[threadIdx.x] = v; __syncthreads();
    for (int w = 128; w > 0; w >>= 1) { if (threadIdx.x < w) sred[threadIdx.x] += sred[threadIdx.x + w]; __syncthreads(); }
    float inv = rsqrtf(sred[0] * (1.0f / DM) + 1e-5f);
    for (int i = threadIdx.x; i < DM; i += 256)
        out[(size_t)row * DM + i] = (_Float16)((x[i] - mean) * inv * g[i] + b[i]);
}

// z = gelu(y_gemm + d_skip*fx) + fx, where fx = LN(h) recomputed in fp32
__global__ __launch_bounds__(256)
void s5_out_gelu(const float* __restrict__ H, const float* __restrict__ Y,
                 const float* __restrict__ dskip,
                 const float* __restrict__ g1, const float* __restrict__ b1,
                 float* __restrict__ Z)
{
    __shared__ float sred[256];
    const int row = blockIdx.x;
    const float* x = H + (size_t)row * DM;
    float s = 0.0f;
    for (int i = threadIdx.x; i < DM; i += 256) s += x[i];
    sred[threadIdx.x] = s; __syncthreads();
    for (int w = 128; w > 0; w >>= 1) { if (threadIdx.x < w) sred[threadIdx.x] += sred[threadIdx.x + w]; __syncthreads(); }
    float mean = sred[0] * (1.0f / DM);
    __syncthreads();
    float v = 0.0f;
    for (int i = threadIdx.x; i < DM; i += 256) { float d = x[i] - mean; v += d * d; }
    sred[threadIdx.x] = v; __syncthreads();
    for (int w = 128; w > 0; w >>= 1) { if (threadIdx.x < w) sred[threadIdx.x] += sred[threadIdx.x + w]; __syncthreads(); }
    float inv = rsqrtf(sred[0] * (1.0f / DM) + 1e-5f);
    for (int i = threadIdx.x; i < DM; i += 256) {
        float fx = (x[i] - mean) * inv * g1[i] + b1[i];
        float y  = Y[(size_t)row * DM + i] + dskip[i] * fx;
        Z[(size_t)row * DM + i] = gelu_exact(y) + fx;
    }
}

// GEGLU elementwise: v = fp16( u[:, :DM] * gelu(u[:, DM:2DM]) )
__global__ void geglu(const float* __restrict__ U, _Float16* __restrict__ V)
{
    int i = blockIdx.x * blockDim.x + threadIdx.x;  // over MTOK*DM
    int m = i >> 10, n = i & (DM - 1);
    float a = U[(size_t)m * DM2 + n];
    float g = U[(size_t)m * DM2 + DM + n];
    V[i] = (_Float16)(a * gelu_exact(g));
}

// h_new = LN( g2 + LN(z; fg,fb) + h ; ng,nb )   (fy recomputed in fp32)
__global__ __launch_bounds__(256)
void ffn_residual_norm(float* __restrict__ H, const float* __restrict__ Z,
                       const float* __restrict__ G2,
                       const float* __restrict__ fg, const float* __restrict__ fb,
                       const float* __restrict__ ng, const float* __restrict__ nb)
{
    __shared__ float sred[256];
    __shared__ float t[DM];
    const int row = blockIdx.x;
    const float* z = Z + (size_t)row * DM;
    float s = 0.0f;
    for (int i = threadIdx.x; i < DM; i += 256) s += z[i];
    sred[threadIdx.x] = s; __syncthreads();
    for (int w = 128; w > 0; w >>= 1) { if (threadIdx.x < w) sred[threadIdx.x] += sred[threadIdx.x + w]; __syncthreads(); }
    float mean = sred[0] * (1.0f / DM);
    __syncthreads();
    float v = 0.0f;
    for (int i = threadIdx.x; i < DM; i += 256) { float d = z[i] - mean; v += d * d; }
    sred[threadIdx.x] = v; __syncthreads();
    for (int w = 128; w > 0; w >>= 1) { if (threadIdx.x < w) sred[threadIdx.x] += sred[threadIdx.x + w]; __syncthreads(); }
    float inv = rsqrtf(sred[0] * (1.0f / DM) + 1e-5f);
    __syncthreads();
    for (int i = threadIdx.x; i < DM; i += 256) {
        float fy = (z[i] - mean) * inv * fg[i] + fb[i];
        t[i] = G2[(size_t)row * DM + i] + fy + H[(size_t)row * DM + i];
    }
    __syncthreads();
    float s2 = 0.0f;
    for (int i = threadIdx.x; i < DM; i += 256) s2 += t[i];
    sred[threadIdx.x] = s2; __syncthreads();
    for (int w = 128; w > 0; w >>= 1) { if (threadIdx.x < w) sred[threadIdx.x] += sred[threadIdx.x + w]; __syncthreads(); }
    float mean2 = sred[0] * (1.0f / DM);
    __syncthreads();
    float v2 = 0.0f;
    for (int i = threadIdx.x; i < DM; i += 256) { float d = t[i] - mean2; v2 += d * d; }
    sred[threadIdx.x] = v2; __syncthreads();
    for (int w = 128; w > 0; w >>= 1) { if (threadIdx.x < w) sred[threadIdx.x] += sred[threadIdx.x + w]; __syncthreads(); }
    float inv2 = rsqrtf(sred[0] * (1.0f / DM) + 1e-5f);
    for (int i = threadIdx.x; i < DM; i += 256)
        H[(size_t)row * DM + i] = (t[i] - mean2) * inv2 * ng[i] + nb[i];
}

// mean-pool over L then dot with dec_w + dec_b. One block per batch.
__global__ __launch_bounds__(256)
void pool_decode(const float* __restrict__ h, const float* __restrict__ dec_w,
                 const float* __restrict__ dec_b, float* __restrict__ out)
{
    __shared__ float sred[256];
    const int b = blockIdx.x;
    const float* hb = h + (size_t)b * SEQ * DM;
    float acc = 0.0f;
    for (int i = threadIdx.x; i < SEQ * DM; i += 256)
        acc += hb[i] * dec_w[i & (DM - 1)];
    sred[threadIdx.x] = acc; __syncthreads();
    for (int w = 128; w > 0; w >>= 1) { if (threadIdx.x < w) sred[threadIdx.x] += sred[threadIdx.x + w]; __syncthreads(); }
    if (threadIdx.x == 0) out[b] = sred[0] * (1.0f / SEQ) + dec_b[0];
}

// ---------------------------------------------------------------------------
extern "C" void kernel_launch(void* const* d_in, const int* in_sizes, int n_in,
                              void* d_out, int out_size, void* d_ws, size_t ws_size,
                              hipStream_t stream)
{
    (void)in_sizes; (void)n_in; (void)out_size; (void)ws_size;

    const float* x        = (const float*)d_in[0];
    const float* enc_w    = (const float*)d_in[1];
    const float* enc_b    = (const float*)d_in[2];
    const float* ln1_g    = (const float*)d_in[3];
    const float* ln1_b    = (const float*)d_in[4];
    const float* lam_re   = (const float*)d_in[5];
    const float* lam_im   = (const float*)d_in[6];
    const float* b_re     = (const float*)d_in[7];
    const float* b_im     = (const float*)d_in[8];
    const float* c_re     = (const float*)d_in[9];
    const float* c_im     = (const float*)d_in[10];
    const float* d_skip   = (const float*)d_in[11];
    const float* log_step = (const float*)d_in[12];
    const float* ffn_g    = (const float*)d_in[13];
    const float* ffn_b    = (const float*)d_in[14];
    const float* ff_enc_w = (const float*)d_in[15];
    const float* ff_dec_w = (const float*)d_in[16];
    const float* norm_g   = (const float*)d_in[17];
    const float* norm_b   = (const float*)d_in[18];
    const float* dec_w    = (const float*)d_in[19];
    const float* dec_b    = (const float*)d_in[20];
    float* out            = (float*)d_out;

    // workspace carve-out
    char* ws = (char*)d_ws;
    float*    h32   = (float*)ws;    ws += (size_t)MTOK * DM  * 4;   //  64 MB
    float*    z32   = (float*)ws;    ws += (size_t)MTOK * DM  * 4;   //  64 MB
    float*    big32 = (float*)ws;    ws += (size_t)MTOK * DM2 * 4;   // 128 MB
    _Float16* a16   = (_Float16*)ws; ws += (size_t)MTOK * DM2 * 2;   //  64 MB
    _Float16* wbu   = (_Float16*)ws; ws += (size_t)DM2 * DM * 2;     //   4 MB
    _Float16* wc    = (_Float16*)ws; ws += (size_t)DM * DM2 * 2;     //   4 MB
    _Float16* w1t   = (_Float16*)ws; ws += (size_t)DM2 * DM * 2;     //   4 MB
    _Float16* w2t   = (_Float16*)ws; ws += (size_t)DM * DM * 2;      //   2 MB
    _Float16* ewt   = (_Float16*)ws; ws += (size_t)DM * DIN * 2;     // 128 KB
    float*    sr    = (float*)ws;    ws += (size_t)NL * DM * 4;
    float*    si    = (float*)ws;    ws += (size_t)NL * DM * 4;
    float*    lbr   = (float*)ws;    ws += (size_t)NL * DM * 4;
    float*    lbi   = (float*)ws;    ws += (size_t)NL * DM * 4;
    float*    car_r = (float*)ws;    ws += (size_t)BATCH * CH * DM * 4;  // 512 KB
    float*    car_i = (float*)ws;    ws += (size_t)BATCH * CH * DM * 4;  // 512 KB

    // derived SSM params (all layers at once)
    ssm_params<<<(NL * DM) / 256, 256, 0, stream>>>(lam_re, lam_im, log_step, sr, si, lbr, lbi);

    // encoder: h = x @ enc_w + enc_b   (K=64)
    pack_transpose<<<(DIN * DM) / 256, 256, 0, stream>>>(enc_w, ewt, DIN, DM);
    f32_to_f16<<<(MTOK * DIN) / 256, 256, 0, stream>>>(x, a16, MTOK * DIN);
    gemm_f16_wmma<<<dim3(MTOK / 128, DM / 128), 256, 0, stream>>>(a16, ewt, h32, enc_b, MTOK, DM, DIN);

    for (int i = 0; i < NL; ++i) {
        const float* br_i = b_re + (size_t)i * DM * DM;
        const float* bi_i = b_im + (size_t)i * DM * DM;
        const float* cr_i = c_re + (size_t)i * DM * DM;
        const float* ci_i = c_im + (size_t)i * DM * DM;
        const float* w1_i = ff_enc_w + (size_t)i * DM * DM2;
        const float* w2_i = ff_dec_w + (size_t)i * DM * DM;

        // per-layer weight packing (fp16, [N,K] layout)
        pack_bbar<<<(DM * DM) / 256, 256, 0, stream>>>(br_i, bi_i, sr + i * DM, si + i * DM, wbu);
        pack_c<<<(DM * DM) / 256, 256, 0, stream>>>(cr_i, ci_i, wc);
        pack_transpose<<<(DM * DM2) / 256, 256, 0, stream>>>(w1_i, w1t, DM, DM2);
        pack_transpose<<<(DM * DM) / 256, 256, 0, stream>>>(w2_i, w2t, DM, DM);

        // fx = LN(h) -> fp16
        ln_to_f16<<<MTOK, 256, 0, stream>>>(h32, ln1_g + i * DM, ln1_b + i * DM, a16);
        // Bu (re|im) = fx @ Bbar^T : [M, 2048]
        gemm_f16_wmma<<<dim3(MTOK / 128, DM2 / 128), 256, 0, stream>>>(a16, wbu, big32, nullptr, MTOK, DM2, DM);
        // chunked complex scan in-place
        scan_chunk<<<(BATCH * CH * DM) / 256, 256, 0, stream>>>(big32, lbr + i * DM, lbi + i * DM, car_r, car_i);
        scan_carry<<<(BATCH * DM) / 256, 256, 0, stream>>>(lbr + i * DM, lbi + i * DM, car_r, car_i);
        scan_apply<<<(BATCH * CH * DM) / 256, 256, 0, stream>>>(big32, lbr + i * DM, lbi + i * DM, car_r, car_i);
        // xs -> fp16 A operand [M, 2048]
        f32_to_f16<<<(MTOK * DM2) / 256, 256, 0, stream>>>(big32, a16, MTOK * DM2);
        // y = Re(xs @ C^T) : one real GEMM K=2048 -> first M*DM of big32
        gemm_f16_wmma<<<dim3(MTOK / 128, DM / 128), 256, 0, stream>>>(a16, wc, big32, nullptr, MTOK, DM, DM2);
        // z = gelu(y + d*fx) + fx
        s5_out_gelu<<<MTOK, 256, 0, stream>>>(h32, big32, d_skip + i * DM,
                                              ln1_g + i * DM, ln1_b + i * DM, z32);
        // fy = LN(z) -> fp16
        ln_to_f16<<<MTOK, 256, 0, stream>>>(z32, ffn_g + i * DM, ffn_b + i * DM, a16);
        // u = fy @ W1 : [M, 2048]
        gemm_f16_wmma<<<dim3(MTOK / 128, DM2 / 128), 256, 0, stream>>>(a16, w1t, big32, nullptr, MTOK, DM2, DM);
        // v = a * gelu(g) -> fp16
        geglu<<<(MTOK * DM) / 256, 256, 0, stream>>>(big32, a16);
        // g2 = v @ W2 : [M, 1024]
        gemm_f16_wmma<<<dim3(MTOK / 128, DM / 128), 256, 0, stream>>>(a16, w2t, big32, nullptr, MTOK, DM, DM);
        // h = LN(g2 + fy + h)
        ffn_residual_norm<<<MTOK, 256, 0, stream>>>(h32, z32, big32,
                                                    ffn_g + i * DM, ffn_b + i * DM,
                                                    norm_g + i * DM, norm_b + i * DM);
    }

    pool_decode<<<BATCH, 256, 0, stream>>>(h32, dec_w, dec_b, out);
}
